// LSTMLosses_66735201846017
// MI455X (gfx1250) — matrix operand order repaired and checked by reading, not compiled, not run on hardware
//
#include <hip/hip_runtime.h>
#include <hip/hip_bf16.h>

// Problem constants (match reference)
#define BB 16
#define TT 512
#define DD 512
#define NMAX (BB*TT + 2*BB)   // 8224 max bank rows (multiple of 32)
#define NT16 (TT/16)          // 32 t-tiles per sequence
#define WG_PER_DIR (BB*NT16)  // 512
#define LDS_STRIDE 520        // 512 + 8 halves -> conflict-free b128 LDS reads

typedef __bf16 bf16_t;
typedef __attribute__((ext_vector_type(16))) __bf16 v16bf;
typedef __attribute__((ext_vector_type(8)))  float  v8f;
typedef __attribute__((ext_vector_type(4)))  float  f32x4;

union V16 { v16bf v; f32x4 f4[2]; };

__device__ __forceinline__ unsigned short f2bf(float f){
  unsigned u = __float_as_uint(f);
  u = u + 0x7FFFu + ((u >> 16) & 1u);   // round-to-nearest-even
  return (unsigned short)(u >> 16);
}

__device__ __forceinline__ void lse_update(float& m, float& s, float v){
  if (v <= m) { s += __expf(v - m); }
  else        { s = s * __expf(m - v) + 1.0f; m = v; }
}

__device__ __forceinline__ void lse_merge(float& m, float& s, float m2, float s2){
  float M = fmaxf(m, m2);
  if (M == -__builtin_inff()) { m = M; s = 0.0f; return; }
  s = s * __expf(m - M) + s2 * __expf(m2 - M);
  m = M;
}

// ---------------------------------------------------------------------------
// Kernel A: scatter feats into the zero-sentinel bank, converted to bf16.
// Row n of the bank: sentinel/padding -> 0, else feats[srcB, srcT, :].
// ---------------------------------------------------------------------------
__global__ void bank_build_kernel(const float* __restrict__ feats,
                                  const int* __restrict__ sl,
                                  unsigned short* __restrict__ xbf){
  const int n = blockIdx.x;
  int cum = 0, srcB = -1, srcT = 0;
  #pragma unroll
  for (int i = 0; i < BB; ++i){
    const int L  = sl[i];
    const int st = 2*i + cum;             // bank start of sequence i
    if (n > st && n <= st + L) { srcB = i; srcT = n - st - 1; }
    cum += L;
  }
  const float* src = (srcB >= 0) ? (feats + ((size_t)srcB*TT + (size_t)srcT)*DD) : nullptr;
  unsigned short* dst = xbf + (size_t)n * DD;
  for (int k = threadIdx.x; k < DD; k += blockDim.x){
    dst[k] = f2bf(src ? src[k] : 0.0f);
  }
}

// ---------------------------------------------------------------------------
// Kernel B: fused GEMM + online logsumexp + diag capture.
// One WG per (dir, b, 16-row t-tile). 4 waves; each wave owns a strided set
// of 32-column bank chunks and keeps per-row streaming softmax state.
// ---------------------------------------------------------------------------
__launch_bounds__(128)
__global__ void logits_lse_kernel(const float* __restrict__ hidden,
                                  const int* __restrict__ sl,
                                  const unsigned short* __restrict__ xbf,
                                  float* __restrict__ partials){
  __shared__ __align__(16) unsigned short As[16 * LDS_STRIDE];
  __shared__ float wmx[4][16], wsm[4][16], wdg[4][16];

  const int wg  = blockIdx.x;
  const int dir = wg >> 9;           // 0 = fw, 1 = bw
  const int rem = wg & 511;
  const int b   = rem >> 5;
  const int t0  = (rem & 31) << 4;
  const int tid = threadIdx.x;

  // Recompute bank layout from seq_lens (16 scalar loads, trivial)
  int cum = 0, start = 0;
  #pragma unroll
  for (int i = 0; i < BB; ++i){ if (i == b) start = 2*i + cum; cum += sl[i]; }
  const int total = cum + 2*BB;
  const int Lb    = sl[b];
  const int pidx  = dir * WG_PER_DIR + rem;

  if (t0 >= Lb){ if (tid == 0) partials[pidx] = 0.0f; return; }

  // Stage A-tile: 16 rows of hidden (one direction slice), f32 -> bf16 in LDS
  const float* hbase = hidden + ((size_t)b*TT + (size_t)t0) * (2*DD) + (size_t)dir*DD;
  for (int idx = tid; idx < 16*DD; idx += 128){
    const int r = idx >> 9, k = idx & (DD - 1);
    As[r*LDS_STRIDE + k] = f2bf(hbase[(size_t)r*(2*DD) + k]);
  }
  __syncthreads();

  const int wave = tid >> 5;
  const int lane = tid & 31;
  const int half = lane >> 4;   // ISA 16-bit layouts split wave into 16-lane halves
  const int lcol = lane & 15;

  // Per-lane row set (C layout: VGPR r -> M = r + 8*half) and diag targets
  int ctab[8];
  #pragma unroll
  for (int r = 0; r < 8; ++r){
    const int t   = t0 + r + 8*half;
    int tgt = dir ? (start + t) : (start + 2 + t);
    ctab[r] = (tgt < total) ? tgt : (total - 1);
  }

  float mrun[8], srun[8], dsum[8];
  #pragma unroll
  for (int r = 0; r < 8; ++r){ mrun[r] = -__builtin_inff(); srun[r] = 0.f; dsum[r] = 0.f; }

  const int nChunks = (total + 31) >> 5;             // 32-column chunks
  // A fragment base: lane holds row M=lcol; K-interleave {0..7,16..23}/{8..15,24..31}
  const unsigned short* Asrow = As + lcol * LDS_STRIDE + half * 8;

  for (int ch = wave; ch < nChunks; ch += 4){
    const int n0 = ch << 5;
    v8f c0 = {}; v8f c1 = {};
    // B fragment: lane holds column n0+lcol (+16 for 2nd tile), K-interleave {0..7,16..23}/{8..15,24..31}
    const unsigned short* bp = xbf + (size_t)(n0 + lcol) * DD + half * 8;
    #pragma unroll
    for (int ks = 0; ks < 16; ++ks){
      const int kb = ks * 32;
      V16 a, b0, b1;
      a.f4[0]  = *reinterpret_cast<const f32x4*>(Asrow + kb);        // K kb+{0..7}/{8..15}
      a.f4[1]  = *reinterpret_cast<const f32x4*>(Asrow + kb + 16);   // K kb+{16..23}/{24..31}
      b0.f4[0] = *reinterpret_cast<const f32x4*>(bp + kb);
      b0.f4[1] = *reinterpret_cast<const f32x4*>(bp + kb + 16);
      b1.f4[0] = *reinterpret_cast<const f32x4*>(bp + (size_t)16*DD + kb);
      b1.f4[1] = *reinterpret_cast<const f32x4*>(bp + (size_t)16*DD + kb + 16);
      c0 = __builtin_amdgcn_wmma_f32_16x16x32_bf16(false, a.v, false, b0.v,
                                                   (short)0, c0, false, false);
      c1 = __builtin_amdgcn_wmma_f32_16x16x32_bf16(false, a.v, false, b1.v,
                                                   (short)0, c1, false, false);
    }
    __builtin_prefetch(bp + (size_t)128 * DD, 0, 1);  // this wave's next chunk (+4*32 rows)

    // Streaming logsumexp update + diag capture on the 16x16 tiles
    const int nA = n0 + lcol, nB = n0 + 16 + lcol;
    const bool vA = nA < total, vB = nB < total;
    #pragma unroll
    for (int r = 0; r < 8; ++r){
      if (vA){ lse_update(mrun[r], srun[r], c0[r]); if (nA == ctab[r]) dsum[r] += c0[r]; }
      if (vB){ lse_update(mrun[r], srun[r], c1[r]); if (nB == ctab[r]) dsum[r] += c1[r]; }
    }
  }

  // Merge across the 16 column-lanes within each half-wave
  #pragma unroll
  for (int off = 1; off < 16; off <<= 1){
    #pragma unroll
    for (int r = 0; r < 8; ++r){
      const float m2 = __shfl_xor(mrun[r], off, 16);
      const float s2 = __shfl_xor(srun[r], off, 16);
      const float d2 = __shfl_xor(dsum[r], off, 16);
      lse_merge(mrun[r], srun[r], m2, s2);
      dsum[r] += d2;
    }
  }

  if (lcol == 0){
    #pragma unroll
    for (int r = 0; r < 8; ++r){
      const int row = r + 8*half;
      wmx[wave][row] = mrun[r]; wsm[wave][row] = srun[r]; wdg[wave][row] = dsum[r];
    }
  }
  __syncthreads();

  if (tid == 0){
    const float invw = 1.0f / ((float)Lb * (float)BB);
    float acc = 0.f;
    for (int row = 0; row < 16; ++row){
      float m = -__builtin_inff(), s = 0.f, d = 0.f;
      #pragma unroll
      for (int w = 0; w < 4; ++w){ lse_merge(m, s, wmx[w][row], wsm[w][row]); d += wdg[w][row]; }
      const int t = t0 + row;
      if (t < Lb) acc += (m + __logf(s) - d) * invw;   // -(logit - lse)/(L*B)
    }
    partials[pidx] = acc;
  }
}

// ---------------------------------------------------------------------------
// Kernel C: deterministic serial reduction of per-WG partials -> 2 losses.
// ---------------------------------------------------------------------------
__global__ void final_reduce_kernel(const float* __restrict__ partials,
                                    float* __restrict__ out){
  const int d = threadIdx.x;
  if (d < 2){
    float s = 0.f;
    for (int i = 0; i < WG_PER_DIR; ++i) s += partials[(size_t)d*WG_PER_DIR + i];
    out[d] = s;
  }
}

extern "C" void kernel_launch(void* const* d_in, const int* in_sizes, int n_in,
                              void* d_out, int out_size, void* d_ws, size_t ws_size,
                              hipStream_t stream){
  const float* feats  = (const float*)d_in[0];
  const float* hidden = (const float*)d_in[1];
  const int*   sl     = (const int*)d_in[2];

  unsigned short* xbf = (unsigned short*)d_ws;                       // 8224*512*2 B
  float* partials = (float*)((char*)d_ws + (size_t)NMAX * DD * sizeof(unsigned short));

  bank_build_kernel<<<NMAX, 256, 0, stream>>>(feats, sl, xbf);
  logits_lse_kernel<<<2 * WG_PER_DIR, 128, 0, stream>>>(hidden, sl, xbf, partials);
  final_reduce_kernel<<<1, 64, 0, stream>>>(partials, (float*)d_out);

  (void)in_sizes; (void)n_in; (void)out_size; (void)ws_size;
}
